// MoE_mod_67224828117003
// MI455X (gfx1250) — matrix operand order repaired and checked by reading, not compiled, not run on hardware
//
#include <hip/hip_runtime.h>
#include <hip/hip_bf16.h>

typedef __attribute__((ext_vector_type(16))) _Float16 v16h;
typedef __attribute__((ext_vector_type(8)))  _Float16 v8h;
typedef __attribute__((ext_vector_type(8)))  float    v8f;
typedef __attribute__((ext_vector_type(4)))  unsigned int v4u;
typedef __attribute__((ext_vector_type(8)))  int      v8i;
typedef __attribute__((ext_vector_type(4)))  int      v4i;

#define B_   128
#define C_   256
#define HW_  1024
#define E_   16
#define K_   4
#define CHW_ (C_ * HW_)   // 262144
#define EPSV 2.220446049250313e-16f

// gating K-split
#define GKB_   256                 // #K-blocks
#define GCHUNK (CHW_ / GKB_)       // 1024 K per block
#define GSW    1032                // LDS row stride (halves) for w_gate chunk

static __device__ __forceinline__ v16h make16(v8h lo, v8h hi) {
    return __builtin_shufflevector(lo, hi, 0,1,2,3,4,5,6,7,8,9,10,11,12,13,14,15);
}
static __device__ __forceinline__ v8h cvt8(float4 a, float4 b) {
    v8h r;
    r[0]=(_Float16)a.x; r[1]=(_Float16)a.y; r[2]=(_Float16)a.z; r[3]=(_Float16)a.w;
    r[4]=(_Float16)b.x; r[5]=(_Float16)b.y; r[6]=(_Float16)b.z; r[7]=(_Float16)b.w;
    return r;
}

// ---------------------------------------------------------------------------
// Phase 1a: WMMA gating partials. Grid = GKB_ K-blocks; block = 256 thr (8 waves).
// Wave w owns batch-tile [w*16, w*16+16); accumulates a 16(batch) x 16(expert)
// f32 tile over its 1024-deep K chunk. w_gate chunk staged once in LDS (f16,
// transposed [e][k]) so w_gate is read from HBM exactly once across the grid.
// ---------------------------------------------------------------------------
__global__ __launch_bounds__(256) void gate_wmma_kernel(
    const float* __restrict__ x, const float* __restrict__ wg,
    float* __restrict__ partials) {

    const int kb = blockIdx.x;
    const int k0 = kb * GCHUNK;

    __shared__ __align__(16) _Float16 shw[E_ * GSW];   // [e][k], padded rows

    {   // stage w_gate chunk: 1024 k x 16 e floats, coalesced float4 reads
        const float4* wsrc = (const float4*)(wg + (size_t)k0 * E_);
        const int tid = threadIdx.x;
#pragma unroll
        for (int j = 0; j < 16; ++j) {
            const int f4 = j * 256 + tid;          // 0..4095
            float4 v = wsrc[f4];
            const int k = f4 >> 2;
            const int e = (f4 & 3) * 4;
            shw[(e + 0) * GSW + k] = (_Float16)v.x;
            shw[(e + 1) * GSW + k] = (_Float16)v.y;
            shw[(e + 2) * GSW + k] = (_Float16)v.z;
            shw[(e + 3) * GSW + k] = (_Float16)v.w;
        }
    }
    __syncthreads();

    const int lane = threadIdx.x & 31;
    const int wave = threadIdx.x >> 5;
    const int hi   = (lane >= 16) ? 1 : 0;
    const int m    = lane & 15;
    const int b0   = wave * 16;

    v8f acc = (v8f){};
    const _Float16* shrow = &shw[m * GSW];             // row e = lane%16

    for (int kk = 0; kk < GCHUNK; kk += 32) {
        const v8h* bp = (const v8h*)(shrow + kk + (hi ? 8 : 0));
        v16h bfrag = make16(bp[0], bp[2]);
        const float4* pa = (const float4*)
            (x + (size_t)(b0 + m) * CHW_ + k0 + kk + (hi ? 8 : 0));
        v16h afrag = make16(cvt8(pa[0], pa[1]), cvt8(pa[4], pa[5]));
        acc = __builtin_amdgcn_wmma_f32_16x16x32_f16(
            false, afrag, false, bfrag, (short)0, acc, false, false);
    }

#pragma unroll
    for (int i = 0; i < 8; ++i) {
        const int b = b0 + i + (hi ? 8 : 0);
        partials[((size_t)kb * B_ + b) * E_ + m] = acc[i];
    }
}

// Phase 1b: deterministic reduce over the K-blocks
__global__ __launch_bounds__(256) void gate_reduce_kernel(
    const float* __restrict__ partials, float* __restrict__ logits) {
    const int t = blockIdx.x * 256 + threadIdx.x;      // 0..2047 = b*16+e
    if (t >= B_ * E_) return;
    float s = 0.f;
    for (int kb = 0; kb < GKB_; ++kb) s += partials[(size_t)kb * (B_ * E_) + t];
    logits[t] = s;
}

// ---------------------------------------------------------------------------
// Phase 1c: wT[e][o][c] = (f16) w_exp[e][c][o]
// ---------------------------------------------------------------------------
__global__ __launch_bounds__(256) void wtrans_kernel(const float* __restrict__ wexp,
                                                     _Float16* __restrict__ wT) {
    const int eo = blockIdx.x;          // e*256 + o
    const int c  = threadIdx.x;
    const int e  = eo >> 8, o = eo & 255;
    wT[(size_t)eo * C_ + c] = (_Float16)wexp[((size_t)e * C_ + c) * C_ + o];
}

// ---------------------------------------------------------------------------
// Phase 1d: xT[b][hw][c] = (f16) x[b][c][hw]  (32x32 LDS tile transpose)
// ---------------------------------------------------------------------------
__global__ __launch_bounds__(256) void xtrans_kernel(const float* __restrict__ x,
                                                     _Float16* __restrict__ xT) {
    __shared__ _Float16 t[32][33];
    const int b = blockIdx.z, c0 = blockIdx.y * 32, hw0 = blockIdx.x * 32;
    const int hx = threadIdx.x & 31, gy = threadIdx.x >> 5;   // gy 0..7
#pragma unroll
    for (int r = 0; r < 4; ++r) {
        const int c = gy * 4 + r;
        t[c][hx] = (_Float16)x[((size_t)b * C_ + c0 + c) * HW_ + hw0 + hx];
    }
    __syncthreads();
#pragma unroll
    for (int r = 0; r < 4; ++r) {
        const int hw = gy * 4 + r;
        xT[((size_t)b * HW_ + hw0 + hw) * C_ + c0 + hx] = t[hx][hw];
    }
}

// ---------------------------------------------------------------------------
// Phase 2: top-4 (lowest-index tie break), softmax, gates output
// ---------------------------------------------------------------------------
__global__ void topk_kernel(const float* __restrict__ logits,
                            float* __restrict__ gk, int* __restrict__ idx,
                            float* __restrict__ gates) {
    const int b = threadIdx.x;
    if (b >= B_) return;
    float l[E_];
#pragma unroll
    for (int e = 0; e < E_; ++e) l[e] = logits[b * E_ + e];
    bool used[E_];
#pragma unroll
    for (int e = 0; e < E_; ++e) used[e] = false;
    int   id[K_];
    float v[K_];
#pragma unroll
    for (int t = 0; t < K_; ++t) {
        float best = -3.402823466e+38f; int bi = 0;
        for (int e = 0; e < E_; ++e)
            if (!used[e] && l[e] > best) { best = l[e]; bi = e; }
        used[bi] = true; id[t] = bi; v[t] = best;
    }
    const float m = v[0];
    float p[K_], s = 0.f;
#pragma unroll
    for (int t = 0; t < K_; ++t) { p[t] = __expf(v[t] - m); s += p[t]; }
    const float inv = 1.f / s;
    float g[E_];
#pragma unroll
    for (int e = 0; e < E_; ++e) g[e] = 0.f;
#pragma unroll
    for (int t = 0; t < K_; ++t) {
        p[t] *= inv;
        gk[b * K_ + t]  = p[t];
        idx[b * K_ + t] = id[t];
        g[id[t]] = p[t];
    }
#pragma unroll
    for (int e = 0; e < E_; ++e) gates[b * E_ + e] = g[e];
}

// ---------------------------------------------------------------------------
// Phase 3: per (b, 64-wide hw tile): y = log(sum_k gk * exp(W_k^T X + b_k))
// 512 threads = 16 waves; wave -> 1 o-tile; 4 hw-tiles x 4 experts of v8f acc.
// B-operand tile (64 hw rows x 256 c, f16) is DMA'd into LDS by the Tensor
// Data Mover with descriptor-driven row padding (264-half stride), then read
// as ds_load_b128 fragments. A fragments stream from L2-resident wT.
// ---------------------------------------------------------------------------
#define LDSROW 264   // 256 + 8 halves of padding (TDM pad_amount)
#define HWB    64    // hw columns per block

__global__ __launch_bounds__(512) void moe_expert_kernel(
    const _Float16* __restrict__ xT, const _Float16* __restrict__ wT,
    const float* __restrict__ b_exp, const float* __restrict__ gk,
    const int* __restrict__ idx, float* __restrict__ y) {

    const int b   = blockIdx.y;
    const int hw0 = blockIdx.x * HWB;

    __shared__ __align__(16) _Float16 sh[HWB * LDSROW];   // X^T : [n(hw)][c]

    // ---- TDM: one wave issues the 64x256 f16 tile load (TENSORcnt tracked)
    if (threadIdx.x < 32) {
        const unsigned lds_base = (unsigned)(size_t)(void*)&sh[0];
        const unsigned long long ga =
            (unsigned long long)(size_t)(xT + ((size_t)b * HW_ + hw0) * C_);
        v4u g0;
        g0[0] = 1u;                                    // count=1, load, no gather
        g0[1] = lds_base;                              // lds_addr
        g0[2] = (unsigned)ga;                          // global_addr[31:0]
        g0[3] = (unsigned)((ga >> 32) & 0x01FFFFFFu) | (2u << 30);  // addr[56:32], type=2
        v8i g1;
        g1[0] = 0x07910000;  // wg_mask=0, data_size=1(2B), pad_en=1, pad_int=6(128dw), pad_amt=3(4dw)
        g1[1] = 0x01000000;  // tensor_dim0 = 256 (c)
        g1[2] = 0x04000000;  // tensor_dim1 = 1024 (hw rows)
        g1[3] = 0x01000000;  // tile_dim0 = 256
        g1[4] = 64;          // tile_dim1 = 64, tile_dim2 = 0
        g1[5] = 256;         // tensor_dim0_stride = 256
        g1[6] = 0;
        g1[7] = 0;
        v4i gz4 = (v4i){0, 0, 0, 0};
        v8i gz8 = (v8i){0, 0, 0, 0, 0, 0, 0, 0};
        __builtin_amdgcn_tensor_load_to_lds(g0, g1, gz4, gz4, gz8, 0);
        __builtin_amdgcn_s_wait_tensorcnt(0);
    }
    __syncthreads();

    const int lane = threadIdx.x & 31;
    const int wave = threadIdx.x >> 5;                    // 0..15
    const int hi   = (lane >= 16) ? 1 : 0;
    const int m    = lane & 15;
    const int o0   = wave * 16;

    float gkv[K_]; int eidx[K_];
#pragma unroll
    for (int t = 0; t < K_; ++t) { gkv[t] = gk[b * K_ + t]; eidx[t] = idx[b * K_ + t]; }

    v8f acc[4][K_];
#pragma unroll
    for (int ht = 0; ht < 4; ++ht)
#pragma unroll
        for (int t = 0; t < K_; ++t) acc[ht][t] = (v8f){};

    for (int kk = 0; kk < C_; kk += 32) {
#pragma unroll
        for (int t = 0; t < K_; ++t) {
            const _Float16* arow =
                wT + ((size_t)(eidx[t] * C_ + o0 + m)) * C_ + kk + (hi ? 8 : 0);
            v16h afrag = make16(*(const v8h*)arow, *(const v8h*)(arow + 16));
#pragma unroll
            for (int ht = 0; ht < 4; ++ht) {
                const v8h* bp = (const v8h*)
                    (&sh[(ht * 16 + m) * LDSROW] + kk + (hi ? 8 : 0));
                v16h bfrag = make16(bp[0], bp[2]);
                acc[ht][t] = __builtin_amdgcn_wmma_f32_16x16x32_f16(
                    false, afrag, false, bfrag, (short)0, acc[ht][t], false, false);
            }
        }
    }

    // epilogue: bias, gk*exp, combine, log, store
#pragma unroll
    for (int ht = 0; ht < 4; ++ht) {
        v8f comb = (v8f){};
#pragma unroll
        for (int t = 0; t < K_; ++t) {
            const float* bb = b_exp + (size_t)eidx[t] * C_;
#pragma unroll
            for (int i = 0; i < 8; ++i) {
                const int o = o0 + i + (hi ? 8 : 0);
                comb[i] += gkv[t] * __expf(acc[ht][t][i] + bb[o]);
            }
        }
#pragma unroll
        for (int i = 0; i < 8; ++i) {
            const int o = o0 + i + (hi ? 8 : 0);
            float cc = comb[i];
            cc = (cc == 0.f) ? EPSV : cc;
            y[((size_t)b * C_ + o) * HW_ + hw0 + ht * 16 + m] = __logf(cc);
        }
    }
}

// ---------------------------------------------------------------------------
extern "C" void kernel_launch(void* const* d_in, const int* in_sizes, int n_in,
                              void* d_out, int out_size, void* d_ws, size_t ws_size,
                              hipStream_t stream) {
    (void)in_sizes; (void)n_in; (void)out_size; (void)ws_size;
    const float* x      = (const float*)d_in[0];
    const float* w_gate = (const float*)d_in[1];
    const float* w_exp  = (const float*)d_in[2];
    const float* b_exp  = (const float*)d_in[3];
    // d_in[4] = k (device scalar) : fixed at 4 per setup_inputs

    char* ws = (char*)d_ws;
    float*    logits   = (float*)(ws);                       //  8 KB
    float*    gk       = (float*)(ws + 8192);                //  2 KB
    int*      idx      = (int*)  (ws + 10240);               //  2 KB
    float*    partials = (float*)(ws + 16384);               //  2 MB
    _Float16* wT       = (_Float16*)(ws + 16384 + (size_t)GKB_ * B_ * E_ * 4);  // 2 MB
    _Float16* xT       = (_Float16*)(ws + 16384 + (size_t)GKB_ * B_ * E_ * 4
                                        + (size_t)E_ * C_ * C_ * 2);            // 64 MB

    float* y     = (float*)d_out;                            // 128*256*1024
    float* gates = y + (size_t)B_ * CHW_;                    // 128*16

    gate_wmma_kernel  <<<GKB_, 256, 0, stream>>>(x, w_gate, partials);
    gate_reduce_kernel<<<(B_ * E_ + 255) / 256, 256, 0, stream>>>(partials, logits);
    wtrans_kernel     <<<E_ * C_, 256, 0, stream>>>(w_exp, wT);
    xtrans_kernel     <<<dim3(HW_ / 32, C_ / 32, B_), 256, 0, stream>>>(x, xT);
    topk_kernel       <<<1, 128, 0, stream>>>(logits, gk, idx, gates);
    moe_expert_kernel <<<dim3(HW_ / HWB, B_), 512, 0, stream>>>(xT, wT, b_exp, gk, idx, y);
}
